// POTO_6665789243587
// MI455X (gfx1250) — matrix operand order repaired and checked by reading, not compiled, not run on hardware
//
#include <hip/hip_runtime.h>
#include <hip/hip_bf16.h>
#include <math.h>

typedef unsigned short ushort_t;
typedef unsigned int   uint_t;
typedef __attribute__((ext_vector_type(16))) __bf16 v16bf_t;
typedef __attribute__((ext_vector_type(8)))  float  v8f_t;

union Frag { uint_t u[8]; v16bf_t v; };

__device__ __forceinline__ ushort_t f2bf(float f) {
  uint_t u = __float_as_uint(f);
  u += 0x7FFFu + ((u >> 16) & 1u);          // round-to-nearest-even
  return (ushort_t)(u >> 16);
}

// ---------------------------------------------------------------------------
// Layout converters
// ---------------------------------------------------------------------------
__global__ void convert_x_kernel(const float* __restrict__ in, ushort_t* __restrict__ out,
                                 int HW, int total) {
  int t = blockIdx.x * blockDim.x + threadIdx.x;
  if (t >= total) return;
  int c = t & 255;
  int r = t >> 8;                 // img*HW + hw
  int img = r / HW;
  int hw  = r - img * HW;
  out[t] = f2bf(in[((size_t)img * 256 + c) * HW + hw]);   // NCHW -> NHWC bf16
}

__global__ void convert_w_kernel(const float* __restrict__ w, ushort_t* __restrict__ wt,
                                 int Oreal, int total /* Opad*2304 */) {
  int t = blockIdx.x * blockDim.x + threadIdx.x;
  if (t >= total) return;
  int o = t / 2304;
  int k = t - o * 2304;
  int tap = k >> 8;               // ky*3+kx
  int c   = k & 255;
  float v = (o < Oreal) ? w[((size_t)o * 256 + c) * 9 + tap] : 0.f;
  wt[t] = f2bf(v);                // [O_pad][tap*256+c]
}

// ---------------------------------------------------------------------------
// Implicit-GEMM 3x3 conv with WMMA bf16 -> f32
//   block tile: 128 spatial (M) x 64 out-ch (N), K-step 64, 8 waves
//   wave tile : 32 x 32 (4 accumulators), 8 WMMA per K-step
// ---------------------------------------------------------------------------
__global__ __launch_bounds__(256)
void conv3x3_wmma_kernel(const ushort_t* __restrict__ xin,   // [N][H][W][256] bf16
                         const ushort_t* __restrict__ wt,    // [Opad][2304]   bf16
                         const float*    __restrict__ bias,  // [Oreal]
                         float*          __restrict__ yout,  // [N][HW][Opad]  f32
                         int Hd, int Wd, int HW, int Oreal, int Opad) {
  __shared__ ushort_t Alds[128 * 64];   // 16 KB
  __shared__ ushort_t Blds[64 * 64];    //  8 KB

  const int tid  = threadIdx.x;
  const int lane = tid & 31;
  const int wave = tid >> 5;
  const int wm   = wave & 3;          // 4 waves along M  (32 rows each)
  const int wn   = wave >> 2;         // 2 waves along N  (32 cols each)
  const int img  = blockIdx.z;
  const int hw0  = blockIdx.x * 128;
  const int o0   = blockIdx.y * 64;

  // A staging: 2 threads per row, 32 cols each (4 x uint4)
  const int arow = tid >> 1;          // 0..127
  const int ahalf = tid & 1;          // col half
  const int mg_l = hw0 + arow;
  int yy = 0, xx = 0;
  const bool mvalid = (mg_l < HW);
  if (mvalid) { yy = mg_l / Wd; xx = mg_l - yy * Wd; }
  // B staging: 4 threads per row, 16 cols each (2 x uint4)
  const int brow = tid >> 2;          // 0..63
  const int bseg = tid & 3;

  const size_t xbase = (size_t)img * HW * 256;

  v8f_t acc00 = {}, acc01 = {}, acc10 = {}, acc11 = {};

  const int h   = lane >> 4;          // lane half
  const int hkA = h * 16;
  const int hkB = h * 16;
  const int nl  = lane & 15;
  const ushort_t* Ar0 = &Alds[(wm * 32 +      nl) * 64];
  const ushort_t* Ar1 = &Alds[(wm * 32 + 16 + nl) * 64];
  const ushort_t* Br0 = &Blds[(wn * 32 +      nl) * 64];
  const ushort_t* Br1 = &Blds[(wn * 32 + 16 + nl) * 64];

  for (int k0 = 0; k0 < 2304; k0 += 64) {
    const int tap = k0 >> 8;                 // 64 | 256 -> single tap per step
    const int c0  = (k0 & 255) + ahalf * 32;
    const int dy  = tap / 3 - 1;
    const int dx  = tap - (tap / 3) * 3 - 1;

    // ---- stage A tile (im2col gather, zero halo) ----
    {
      uint4 av[4] = {make_uint4(0u,0u,0u,0u), make_uint4(0u,0u,0u,0u),
                     make_uint4(0u,0u,0u,0u), make_uint4(0u,0u,0u,0u)};
      if (mvalid) {
        int ys = yy + dy, xs = xx + dx;
        if (ys >= 0 && ys < Hd && xs >= 0 && xs < Wd) {
          const uint4* src = (const uint4*)(xin + xbase + (size_t)(ys * Wd + xs) * 256 + c0);
#pragma unroll
          for (int q = 0; q < 4; ++q) av[q] = src[q];
        }
      }
      uint4* dst = (uint4*)&Alds[arow * 64 + ahalf * 32];
#pragma unroll
      for (int q = 0; q < 4; ++q) dst[q] = av[q];
    }
    // ---- stage B tile (contiguous weight rows) ----
    {
      const uint4* src = (const uint4*)(wt + (size_t)(o0 + brow) * 2304 + k0 + bseg * 16);
      uint4* dst = (uint4*)&Blds[brow * 64 + bseg * 16];
#pragma unroll
      for (int q = 0; q < 2; ++q) dst[q] = src[q];
    }

    __syncthreads();

#pragma unroll
    for (int kk = 0; kk < 64; kk += 32) {
      Frag a0, a1, b0, b1;
#pragma unroll
      for (int j = 0; j < 4; ++j) {          // A: pairs K={16h+2j}, {16h+8+2j}
        a0.u[j]     = *(const uint_t*)&Ar0[kk + hkA + 2 * j];
        a0.u[4 + j] = *(const uint_t*)&Ar0[kk + 8 + hkA + 2 * j];
        a1.u[j]     = *(const uint_t*)&Ar1[kk + hkA + 2 * j];
        a1.u[4 + j] = *(const uint_t*)&Ar1[kk + 8 + hkA + 2 * j];
      }
#pragma unroll
      for (int j = 0; j < 8; ++j) {          // B: pairs K = 16h + 2j
        b0.u[j] = *(const uint_t*)&Br0[kk + hkB + 2 * j];
        b1.u[j] = *(const uint_t*)&Br1[kk + hkB + 2 * j];
      }
      acc00 = __builtin_amdgcn_wmma_f32_16x16x32_bf16(false, a0.v, false, b0.v,
                                                      (short)0, acc00, false, false);
      acc01 = __builtin_amdgcn_wmma_f32_16x16x32_bf16(false, a0.v, false, b1.v,
                                                      (short)0, acc01, false, false);
      acc10 = __builtin_amdgcn_wmma_f32_16x16x32_bf16(false, a1.v, false, b0.v,
                                                      (short)0, acc10, false, false);
      acc11 = __builtin_amdgcn_wmma_f32_16x16x32_bf16(false, a1.v, false, b1.v,
                                                      (short)0, acc11, false, false);
    }
    __syncthreads();
  }

  // D layout: VGPR r -> M = r (lanes 0-15) / 8+r (lanes 16-31), N = lane&15
  const int mb = hw0 + wm * 32 + h * 8;
  const int nb = o0 + wn * 32 + nl;
#pragma unroll
  for (int r = 0; r < 8; ++r) {
    int mg0 = mb + r;
    if (mg0 < HW) {
      size_t row = ((size_t)img * HW + mg0) * Opad;
      if (nb      < Oreal) yout[row + nb]      = acc00[r] + bias[nb];
      if (nb + 16 < Oreal) yout[row + nb + 16] = acc01[r] + bias[nb + 16];
    }
    int mg1 = mb + 16 + r;
    if (mg1 < HW) {
      size_t row = ((size_t)img * HW + mg1) * Opad;
      if (nb      < Oreal) yout[row + nb]      = acc10[r] + bias[nb];
      if (nb + 16 < Oreal) yout[row + nb + 16] = acc11[r] + bias[nb + 16];
    }
  }
}

// ---------------------------------------------------------------------------
// GroupNorm (32 groups of 8 channels) stats + apply + ReLU + bf16 repack
// ---------------------------------------------------------------------------
__global__ __launch_bounds__(256)
void gn_stats_kernel(const float* __restrict__ y, float* __restrict__ stat, int HW) {
  __shared__ float s1[256], s2[256];
  const int img = blockIdx.x >> 5;
  const int g   = blockIdx.x & 31;
  float sum = 0.f, sq = 0.f;
  const float* base = y + (size_t)img * HW * 256 + g * 8;
  for (int hw = threadIdx.x; hw < HW; hw += 256) {
    const float* p = base + (size_t)hw * 256;
#pragma unroll
    for (int c = 0; c < 8; ++c) { float v = p[c]; sum += v; sq += v * v; }
  }
  s1[threadIdx.x] = sum; s2[threadIdx.x] = sq;
  __syncthreads();
  for (int off = 128; off > 0; off >>= 1) {
    if (threadIdx.x < off) {
      s1[threadIdx.x] += s1[threadIdx.x + off];
      s2[threadIdx.x] += s2[threadIdx.x + off];
    }
    __syncthreads();
  }
  if (threadIdx.x == 0) {
    float inv  = 1.f / (8.f * (float)HW);
    float mean = s1[0] * inv;
    float var  = s2[0] * inv - mean * mean;
    stat[blockIdx.x * 2]     = mean;
    stat[blockIdx.x * 2 + 1] = rsqrtf(var + 1e-5f);
  }
}

__global__ void gn_apply_kernel(const float* __restrict__ y, const float* __restrict__ stat,
                                const float* __restrict__ gamma, const float* __restrict__ beta,
                                ushort_t* __restrict__ out, int total, int HW) {
  int t = blockIdx.x * blockDim.x + threadIdx.x;
  if (t >= total) return;
  int c   = t & 255;
  int img = (t >> 8) / HW;
  int g   = c >> 3;
  const float* st = stat + ((img << 5) + g) * 2;
  float v = (y[t] - st[0]) * st[1] * gamma[c] + beta[c];
  out[t] = f2bf(fmaxf(v, 0.f));
}

// ---------------------------------------------------------------------------
// Head epilogues
// ---------------------------------------------------------------------------
__global__ void score_kernel(const float* __restrict__ ycls, float* __restrict__ s,
                             int HW, int total) {
  int t = blockIdx.x * blockDim.x + threadIdx.x;
  if (t >= total) return;
  int k      = t % 80;
  int hw_img = t / 80;                       // img*HW + hw
  float v = ycls[(size_t)hw_img * 128 + k];
  s[t] = 1.f / (1.f + __expf(-v));
}

__global__ void delta_kernel(const float* __restrict__ ybox, const float* __restrict__ scales,
                             int lvl, float stride, float* __restrict__ d, int total) {
  int t = blockIdx.x * blockDim.x + threadIdx.x;
  if (t >= total) return;
  int j      = t & 3;
  int hw_img = t >> 2;
  float v = ybox[(size_t)hw_img * 64 + j] * scales[lvl];
  d[t] = fmaxf(v, 0.f) * stride;
}

// ---------------------------------------------------------------------------
// Exact stable top-1000 per (image, level): iterative masked argmax
// ---------------------------------------------------------------------------
__global__ __launch_bounds__(256)
void topk_kernel(float* __restrict__ s, const float* __restrict__ d,
                 int HW, int Wd, float stride, int base,
                 float* __restrict__ cboxes, float* __restrict__ cscores,
                 int* __restrict__ ccls) {
  __shared__ float bv[256];
  __shared__ int   bi[256];
  const int img = blockIdx.x;
  const int len = HW * 80;
  float*       sp = s + (size_t)img * len;
  const float* dp = d + (size_t)img * HW * 4;
  const int tid = threadIdx.x;

  for (int t = 0; t < 1000; ++t) {
    float mv = -1e30f; int mi = 0x7fffffff;
    for (int i = tid; i < len; i += 256) {
      float v = sp[i];
      if (v > mv) { mv = v; mi = i; }        // ascending scan => first tie kept
    }
    bv[tid] = mv; bi[tid] = mi;
    __syncthreads();
    for (int off = 128; off > 0; off >>= 1) {
      if (tid < off) {
        float ov = bv[tid + off]; int oi = bi[tid + off];
        if (ov > bv[tid] || (ov == bv[tid] && oi < bi[tid])) { bv[tid] = ov; bi[tid] = oi; }
      }
      __syncthreads();
    }
    if (tid == 0) {
      int   i = bi[0];
      float v = bv[0];
      int anchor = i / 80, cls = i - anchor * 80;
      int ay = anchor / Wd, ax = anchor - ay * Wd;
      float sx = (ax + 0.5f) * stride, sy = (ay + 0.5f) * stride;
      const float* dd = dp + (size_t)anchor * 4;
      int o = img * 5000 + base + t;
      cboxes[o * 4 + 0] = sx - dd[0];
      cboxes[o * 4 + 1] = sy - dd[1];
      cboxes[o * 4 + 2] = sx + dd[2];
      cboxes[o * 4 + 3] = sy + dd[3];
      cscores[o] = (v > 0.05f) ? v : 0.f;
      ccls[o]    = cls;
      sp[i] = -1.f;                          // mask out
    }
    __syncthreads();
  }
}

// ---------------------------------------------------------------------------
// Class-aware NMS (coordinate-offset trick), 100 selections per image
// ---------------------------------------------------------------------------
__global__ __launch_bounds__(256)
void nms_kernel(const float* __restrict__ cboxes, const float* __restrict__ cscores,
                const int* __restrict__ ccls, float* __restrict__ live,
                float* __restrict__ dout) {
  __shared__ float bv[256];
  __shared__ int   bi[256];
  __shared__ float sel[5];                   // offset box + area
  const int img = blockIdx.x;
  const int tid = threadIdx.x;
  const float* bx = cboxes  + (size_t)img * 5000 * 4;
  const float* sc = cscores + (size_t)img * 5000;
  const int*   cl = ccls    + (size_t)img * 5000;
  float*       lv = live    + (size_t)img * 5000;

  for (int i = tid; i < 5000; i += 256) lv[i] = sc[i];
  __syncthreads();

  for (int t = 0; t < 100; ++t) {
    float mv = -1e30f; int mi = 0x7fffffff;
    for (int i = tid; i < 5000; i += 256) {
      float v = lv[i];
      if (v > mv) { mv = v; mi = i; }
    }
    bv[tid] = mv; bi[tid] = mi;
    __syncthreads();
    for (int off = 128; off > 0; off >>= 1) {
      if (tid < off) {
        float ov = bv[tid + off]; int oi = bi[tid + off];
        if (ov > bv[tid] || (ov == bv[tid] && oi < bi[tid])) { bv[tid] = ov; bi[tid] = oi; }
      }
      __syncthreads();
    }
    const int   i = bi[0];
    const float s = bv[0];
    if (tid == 0) {
      float co = (float)cl[i] * 1e4f;
      sel[0] = bx[i * 4 + 0] + co;
      sel[1] = bx[i * 4 + 1] + co;
      sel[2] = bx[i * 4 + 2] + co;
      sel[3] = bx[i * 4 + 3] + co;
      sel[4] = (sel[2] - sel[0]) * (sel[3] - sel[1]);
      dout[img * 500 + t * 5 + 0] = bx[i * 4 + 0];
      dout[img * 500 + t * 5 + 1] = bx[i * 4 + 1];
      dout[img * 500 + t * 5 + 2] = bx[i * 4 + 2];
      dout[img * 500 + t * 5 + 3] = bx[i * 4 + 3];
      dout[img * 500 + t * 5 + 4] = s;
      dout[1000 + img * 100 + t]  = (float)cl[i];
    }
    __syncthreads();
    if (s > 0.f) {
      float b0 = sel[0], b1 = sel[1], b2 = sel[2], b3 = sel[3], ai = sel[4];
      for (int j = tid; j < 5000; j += 256) {
        float co = (float)cl[j] * 1e4f;
        float j0 = bx[j * 4 + 0] + co, j1 = bx[j * 4 + 1] + co;
        float j2 = bx[j * 4 + 2] + co, j3 = bx[j * 4 + 3] + co;
        float w  = fmaxf(fminf(b2, j2) - fmaxf(b0, j0), 0.f);
        float h2 = fmaxf(fminf(b3, j3) - fmaxf(b1, j1), 0.f);
        float inter = w * h2;
        float aj = (j2 - j0) * (j3 - j1);
        float iou = inter / (ai + aj - inter + 1e-9f);
        if (iou > 0.6f) lv[j] = 0.f;
      }
    }
    __syncthreads();
    if (tid == 0) lv[i] = 0.f;
    __syncthreads();
  }
}

// ---------------------------------------------------------------------------
// Host orchestration
// ---------------------------------------------------------------------------
extern "C" void kernel_launch(void* const* d_in, const int* in_sizes, int n_in,
                              void* d_out, int out_size, void* d_ws, size_t ws_size,
                              hipStream_t stream) {
  (void)in_sizes; (void)n_in; (void)out_size; (void)ws_size;
  static const int   Hs[5]      = {100, 50, 25, 13, 7};
  static const float strides[5] = {8.f, 16.f, 32.f, 64.f, 128.f};
  const int MAXHW = 10000;

  char* ws = (char*)d_ws;
  size_t off = 0;
  auto alloc = [&](size_t bytes) -> char* {
    char* p = ws + off;
    off += (bytes + 255) & ~(size_t)255;
    return p;
  };
  ushort_t* XBUF = (ushort_t*)alloc(2ull * MAXHW * 256 * 2);
  ushort_t* T0   = (ushort_t*)alloc(2ull * MAXHW * 256 * 2);
  ushort_t* T1   = (ushort_t*)alloc(2ull * MAXHW * 256 * 2);
  float*    YF   = (float*)   alloc(2ull * MAXHW * 256 * 4);
  ushort_t* WSUB = (ushort_t*)alloc(8ull * 256 * 2304 * 2);
  ushort_t* WCLS = (ushort_t*)alloc(128ull * 2304 * 2);
  ushort_t* WBOX = (ushort_t*)alloc(64ull  * 2304 * 2);
  float*    STAT = (float*)   alloc(64 * 2 * 4);
  float*    SBUF = (float*)   alloc(2ull * MAXHW * 80 * 4);
  float*    DBUF = (float*)   alloc(2ull * MAXHW * 4 * 4);
  float*    CBOX = (float*)   alloc(2ull * 5000 * 4 * 4);
  float*    CSC  = (float*)   alloc(2ull * 5000 * 4);
  int*      CCLS = (int*)     alloc(2ull * 5000 * 4);
  float*    LIVE = (float*)   alloc(2ull * 5000 * 4);

  // --- weights -> bf16 [O_pad][K] ---
  const int tw = 256 * 2304;
  for (int l = 0; l < 4; ++l) {
    convert_w_kernel<<<(tw + 255) / 256, 256, 0, stream>>>(
        (const float*)d_in[5 + 4 * l],  WSUB + (size_t)l * tw,       256, tw);
    convert_w_kernel<<<(tw + 255) / 256, 256, 0, stream>>>(
        (const float*)d_in[21 + 4 * l], WSUB + (size_t)(4 + l) * tw, 256, tw);
  }
  convert_w_kernel<<<(128 * 2304 + 255) / 256, 256, 0, stream>>>(
      (const float*)d_in[37], WCLS, 80, 128 * 2304);
  convert_w_kernel<<<(64 * 2304 + 255) / 256, 256, 0, stream>>>(
      (const float*)d_in[39], WBOX, 4, 64 * 2304);

  const float* scales = (const float*)d_in[41];
  ushort_t* outs[2] = {T0, T1};

  for (int lvl = 0; lvl < 5; ++lvl) {
    const int H = Hs[lvl], W = Hs[lvl], HW = H * W;
    const int totalX = 2 * HW * 256;

    convert_x_kernel<<<(totalX + 255) / 256, 256, 0, stream>>>(
        (const float*)d_in[lvl], XBUF, HW, totalX);

    dim3 cg((HW + 127) / 128, 1, 2);
    // ---- cls subnet ----
    const ushort_t* in = XBUF;
    for (int l = 0; l < 4; ++l) {
      dim3 g((HW + 127) / 128, 4, 2);
      conv3x3_wmma_kernel<<<g, 256, 0, stream>>>(
          in, WSUB + (size_t)l * tw, (const float*)d_in[6 + 4 * l], YF, H, W, HW, 256, 256);
      gn_stats_kernel<<<64, 256, 0, stream>>>(YF, STAT, HW);
      gn_apply_kernel<<<(totalX + 255) / 256, 256, 0, stream>>>(
          YF, STAT, (const float*)d_in[7 + 4 * l], (const float*)d_in[8 + 4 * l],
          outs[l & 1], totalX, HW);
      in = outs[l & 1];
    }
    {
      dim3 g((HW + 127) / 128, 2, 2);
      conv3x3_wmma_kernel<<<g, 256, 0, stream>>>(
          in, WCLS, (const float*)d_in[38], YF, H, W, HW, 80, 128);
    }
    const int totS = 2 * HW * 80;
    score_kernel<<<(totS + 255) / 256, 256, 0, stream>>>(YF, SBUF, HW, totS);

    // ---- box subnet ----
    in = XBUF;
    for (int l = 0; l < 4; ++l) {
      dim3 g((HW + 127) / 128, 4, 2);
      conv3x3_wmma_kernel<<<g, 256, 0, stream>>>(
          in, WSUB + (size_t)(4 + l) * tw, (const float*)d_in[22 + 4 * l], YF, H, W, HW, 256, 256);
      gn_stats_kernel<<<64, 256, 0, stream>>>(YF, STAT, HW);
      gn_apply_kernel<<<(totalX + 255) / 256, 256, 0, stream>>>(
          YF, STAT, (const float*)d_in[23 + 4 * l], (const float*)d_in[24 + 4 * l],
          outs[l & 1], totalX, HW);
      in = outs[l & 1];
    }
    {
      dim3 g((HW + 127) / 128, 1, 2);
      conv3x3_wmma_kernel<<<g, 256, 0, stream>>>(
          in, WBOX, (const float*)d_in[40], YF, H, W, HW, 4, 64);
    }
    const int totD = 2 * HW * 4;
    delta_kernel<<<(totD + 255) / 256, 256, 0, stream>>>(
        YF, scales, lvl, strides[lvl], DBUF, totD);

    topk_kernel<<<2, 256, 0, stream>>>(SBUF, DBUF, HW, W, strides[lvl],
                                       lvl * 1000, CBOX, CSC, CCLS);
  }

  nms_kernel<<<2, 256, 0, stream>>>(CBOX, CSC, CCLS, LIVE, (float*)d_out);
}
